// Feature2DDecoder_11115375362799
// MI455X (gfx1250) — matrix operand order repaired and checked by reading, not compiled, not run on hardware
//
#include <hip/hip_runtime.h>
#include <cstddef>
#include <cstdint>

// ---------------- types for WMMA / TDM ----------------
typedef __bf16 v16bf __attribute__((ext_vector_type(16)));
typedef __bf16 v8bf  __attribute__((ext_vector_type(8)));
typedef float  v8f   __attribute__((ext_vector_type(8)));
typedef unsigned int u32x4 __attribute__((ext_vector_type(4)));
typedef unsigned int u32x8 __attribute__((ext_vector_type(8)));

union BFrag { v16bf v; v8bf h[2]; };

// ---------------- problem constants ----------------
#define BB   2
#define NN   8192
#define SS   2048
#define DD   256
#define ROWS (BB*NN)       // 16384
#define C1   259
#define K1P  288           // 259 padded to multiple of 32
#define O1   256
#define K2   256
#define O2   256
#define KMAX 288

__device__ __forceinline__ unsigned short f2bf(float f) {
    unsigned int u = __float_as_uint(f);
    unsigned int r = u + 0x7FFFu + ((u >> 16) & 1u);   // round-to-nearest-even
    return (unsigned short)(r >> 16);
}

// -------- weight convert + pad to bf16 (O x Cpad), zeros beyond Cin --------
__global__ void wconv_kernel(const float* __restrict__ src, unsigned short* __restrict__ dst,
                             int O, int Cin, int Cpad) {
    int i = blockIdx.x * 256 + threadIdx.x;
    if (i >= O * Cpad) return;
    int o = i / Cpad, c = i % Cpad;
    dst[i] = (c < Cin) ? f2bf(src[o * Cin + c]) : (unsigned short)0;
}

// -------- 3-NN interpolation; builds Xin (ROWS x K1P) bf16, row-major --------
__global__ void knn_build_kernel(const float* __restrict__ xyz1, const float* __restrict__ xyz2,
                                 const float* __restrict__ p2, unsigned short* __restrict__ Xin) {
    __shared__ float sx[SS], sy[SS], sz[SS];
    int tid = threadIdx.x;
    int b = blockIdx.x / (NN / 256);
    int n = (blockIdx.x % (NN / 256)) * 256 + tid;
    const float* X2 = xyz2 + (size_t)b * 3 * SS;
    for (int i = tid; i < SS; i += 256) { sx[i] = X2[i]; sy[i] = X2[SS + i]; sz[i] = X2[2*SS + i]; }
    __syncthreads();
    const float* X1 = xyz1 + (size_t)b * 3 * NN;
    float px = X1[n], py = X1[NN + n], pz = X1[2*NN + n];
    float d0 = 1e30f, d1 = 1e30f, d2 = 1e30f;
    int i0 = 0, i1 = 0, i2 = 0;
    for (int s = 0; s < SS; ++s) {
        float dx = px - sx[s], dy = py - sy[s], dz = pz - sz[s];
        float d = dx*dx + dy*dy + dz*dz;
        if (d < d0)      { d2 = d1; i2 = i1; d1 = d0; i1 = i0; d0 = d; i0 = s; }
        else if (d < d1) { d2 = d1; i2 = i1; d1 = d;  i1 = s; }
        else if (d < d2) { d2 = d;  i2 = s; }
    }
    float w0 = 1.f/(d0 + 1e-8f), w1 = 1.f/(d1 + 1e-8f), w2 = 1.f/(d2 + 1e-8f);
    float wsum = w0 + w1 + w2; w0 /= wsum; w1 /= wsum; w2 /= wsum;
    size_t row = (size_t)b * NN + n;
    unsigned short* out = Xin + row * K1P;
    out[0] = f2bf(px); out[1] = f2bf(py); out[2] = f2bf(pz);
    const float* P = p2 + (size_t)b * DD * SS;
    for (int c = 0; c < DD; ++c) {
        float v = w0 * P[c*SS + i0] + w1 * P[c*SS + i1] + w2 * P[c*SS + i2];
        out[3 + c] = f2bf(v);
    }
    for (int c = 3 + DD; c < K1P; ++c) out[c] = 0;
}

// -------- density: invd[b*N+n] = 1 / (mean_m exp(-d2/(2 BW^2)) / (2.5 BW)) --------
__global__ void density_kernel(const float* __restrict__ xyz1, float* __restrict__ invd) {
    __shared__ float sx[2048], sy[2048], sz[2048];
    int tid = threadIdx.x;
    int b = blockIdx.x / (NN / 256);
    int n = (blockIdx.x % (NN / 256)) * 256 + tid;
    const float* X1 = xyz1 + (size_t)b * 3 * NN;
    float px = X1[n], py = X1[NN + n], pz = X1[2*NN + n];
    float acc = 0.f;
    for (int t = 0; t < NN; t += 2048) {
        __syncthreads();
        for (int i = tid; i < 2048; i += 256) {
            sx[i] = X1[t + i]; sy[i] = X1[NN + t + i]; sz[i] = X1[2*NN + t + i];
        }
        __syncthreads();
        for (int m = 0; m < 2048; ++m) {
            float dx = px - sx[m], dy = py - sy[m], dz = pz - sz[m];
            acc += __expf(-(dx*dx + dy*dy + dz*dz) * 50.0f);   // 1/(2*0.1^2)=50
        }
    }
    float density = (acc / (float)NN) * (1.0f / 0.25f);        // /(2.5*BW)
    invd[(size_t)b * NN + n] = 1.f / density;
}

__global__ void batch_max_kernel(const float* __restrict__ invd, float* __restrict__ maxv) {
    __shared__ float red[256];
    int b = blockIdx.x, tid = threadIdx.x;
    float m = -1e30f;
    for (int i = tid; i < NN; i += 256) m = fmaxf(m, invd[(size_t)b * NN + i]);
    red[tid] = m; __syncthreads();
    for (int s = 128; s > 0; s >>= 1) { if (tid < s) red[tid] = fmaxf(red[tid], red[tid + s]); __syncthreads(); }
    if (tid == 0) maxv[b] = red[0];
}

__global__ void scale_kernel(const float* __restrict__ invd, const float* __restrict__ maxv,
                             float* __restrict__ s0) {
    int i = blockIdx.x * 256 + threadIdx.x;   // row over ROWS
    int b = i / NN;
    s0[i] = invd[i] / maxv[b];
}

// -------- WMMA bf16 GEMM: out(rows x O) = A(rows x K) * W(O x K)^T + bias --------
// Block = 8 waves covering a 512-row x 16-col tile. The 16xK bf16 weight panel
// is DMA'd into LDS once per block by the Tensor Data Mover (tensor_load_to_lds,
// TENSORcnt), then every wave computes a 64x16 strip: one LDS B-fragment feeds
// four WMMAs (register reuse across 4 row-tiles). A-fragment loads are
// contiguous 16B per the CDNA5 16-bit A(16x32) layout; no transposes anywhere.
__global__ void gemm_bf16_wmma(const unsigned short* __restrict__ A,
                               const unsigned short* __restrict__ Bw,
                               const float* __restrict__ bias,
                               float* __restrict__ out,
                               int rows, int K, int O) {
    __shared__ unsigned short sB[16 * KMAX];
    int lane  = threadIdx.x & 31;
    int wavei = threadIdx.x >> 5;
    int otiles = O >> 4;
    int ot   = blockIdx.x % otiles;
    int grp  = blockIdx.x / otiles;          // 512-row group
    int strip = grp * 8 + wavei;             // this wave's 64-row strip
    int ml = lane & 15, hi = lane >> 4;

    // ---- TDM: DMA 16xK bf16 weight panel (tile = whole 2D tensor) into LDS ----
    if (threadIdx.x < 32) {
        unsigned long long gaddr = (unsigned long long)(uintptr_t)(Bw + (size_t)ot * 16 * K);
        unsigned lds_off = (unsigned)(uintptr_t)(&sB[0]);   // LDS byte offset = low 32 bits
        u32x4 g0;
        g0[0] = 1u;                                          // count=1 valid descriptor
        g0[1] = lds_off;                                     // lds_addr
        g0[2] = (unsigned)gaddr;                             // global_addr[31:0]
        g0[3] = (unsigned)((gaddr >> 32) & 0x1FFFFFFu) | (2u << 30); // addr[56:32] | type=2
        u32x8 g1;
        g1[0] = (1u << 16);                                  // data_size=1 -> 2 bytes
        g1[1] = ((unsigned)K & 0xFFFFu) << 16;               // tensor_dim0[15:0]
        g1[2] = (((unsigned)K >> 16) & 0xFFFFu) | (16u << 16); // tensor_dim0[31:16] | tensor_dim1=16
        g1[3] = ((unsigned)K & 0xFFFFu) << 16;               // tile_dim0 = K
        g1[4] = 16u;                                         // tile_dim1 = 16, tile_dim2 = 0
        g1[5] = (unsigned)K;                                 // tensor_dim0_stride[31:0]
        g1[6] = 0u;                                          // stride hi / dim1_stride lo
        g1[7] = 0u;
        asm volatile("tensor_load_to_lds %0, %1" :: "s"(g0), "s"(g1) : "memory");
        __builtin_amdgcn_s_wait_tensorcnt(0);
    }
    __syncthreads();

    const unsigned short* arow = A + ((size_t)strip * 64 + ml) * K;
    const unsigned short* brow = &sB[ml * K];
    v8f c0 = {}, c1 = {}, c2 = {}, c3 = {};
    for (int kb = 0; kb < K; kb += 32) {
        BFrag b;
        // B 32x16 bf16 from LDS: lanes0-15 K{0..7,16..23} of col, lanes16-31 K{8..15,24..31}
        b.h[0] = *(const v8bf*)(brow + kb + hi * 8);
        b.h[1] = *(const v8bf*)(brow + kb + 16 + hi * 8);
        if (kb + 32 < K) __builtin_prefetch(arow + kb + 32, 0, 1);
        BFrag a;
        // A 16x32 bf16: lanes0-15 K{0..7,16..23}, lanes16-31 K{8..15,24..31}
#define GEMM_STEP(T, CC)                                                     \
        a.h[0] = *(const v8bf*)(arow + (size_t)(T) * 16 * K + kb + hi * 8);  \
        a.h[1] = *(const v8bf*)(arow + (size_t)(T) * 16 * K + kb + 16 + hi * 8); \
        CC = __builtin_amdgcn_wmma_f32_16x16x32_bf16(false, a.v, false, b.v, \
                                                     (short)0, CC, false, false);
        GEMM_STEP(0, c0)
        GEMM_STEP(1, c1)
        GEMM_STEP(2, c2)
        GEMM_STEP(3, c3)
#undef GEMM_STEP
    }
    int ocol = ot * 16 + ml;
    float bv = bias[ocol];
#pragma unroll
    for (int t = 0; t < 4; ++t) {
        v8f c = (t == 0) ? c0 : (t == 1) ? c1 : (t == 2) ? c2 : c3;
        size_t rbase = (size_t)strip * 64 + t * 16 + hi * 8;  // C/D: VGPR r -> row r (+8 hi)
#pragma unroll
        for (int r = 0; r < 8; ++r) out[(rbase + r) * O + ocol] = c[r] + bv;
    }
}

// -------- per-channel batch stats (deterministic tree reduce) --------
__global__ void stats_kernel(const float* __restrict__ x, int rows, int C,
                             float* __restrict__ mean, float* __restrict__ rstd) {
    __shared__ float sa[256], sb[256];
    int c = blockIdx.x, tid = threadIdx.x;
    float a = 0.f, b = 0.f;
    for (int i = tid; i < rows; i += 256) { float v = x[(size_t)i * C + c]; a += v; b += v * v; }
    sa[tid] = a; sb[tid] = b; __syncthreads();
    for (int s = 128; s > 0; s >>= 1) {
        if (tid < s) { sa[tid] += sa[tid + s]; sb[tid] += sb[tid + s]; }
        __syncthreads();
    }
    if (tid == 0) {
        float mu = sa[0] / rows;
        float var = sb[0] / rows - mu * mu;
        mean[c] = mu; rstd[c] = rsqrtf(var + 1e-5f);
    }
}

// mode 0: x = relu(bn(x)) in place; mode 1: obf = bf16(relu(bn(x)));
// mode 2: x = relu(bn(x)) * s[row]
__global__ void bn_relu_kernel(float* __restrict__ x, const float* __restrict__ mean,
                               const float* __restrict__ rstd, const float* __restrict__ g,
                               const float* __restrict__ b, int rows, int C, int mode,
                               unsigned short* __restrict__ obf, const float* __restrict__ s) {
    size_t i = (size_t)blockIdx.x * 256 + threadIdx.x;
    if (i >= (size_t)rows * C) return;
    int c = (int)(i % C); size_t row = i / C;
    float v = (x[i] - mean[c]) * rstd[c] * g[c] + b[c];
    v = fmaxf(v, 0.f);
    if (mode == 1)      obf[i] = f2bf(v);
    else if (mode == 2) x[i] = v * s[row];
    else                x[i] = v;
}

// -------- tiny 1x1 conv for the density/weight branches --------
__global__ void conv_small_kernel(const float* __restrict__ in, const float* __restrict__ W,
                                  const float* __restrict__ cb, float* __restrict__ out,
                                  int rows, int Cin, int Cout) {
    size_t i = (size_t)blockIdx.x * 256 + threadIdx.x;
    if (i >= (size_t)rows * Cout) return;
    int o = (int)(i % Cout); size_t row = i / Cout;
    const float* ir = in + row * Cin;
    const float* wr = W + (size_t)o * Cin;
    float acc = cb[o];
    for (int c = 0; c < Cin; ++c) acc += wr[c] * ir[c];
    out[i] = acc;
}

__global__ void pack_xyz_kernel(const float* __restrict__ xyz1, float* __restrict__ gin) {
    int i = blockIdx.x * 256 + threadIdx.x;   // row
    int b = i / NN, n = i % NN;
    const float* X = xyz1 + (size_t)b * 3 * NN;
    gin[(size_t)i * 3 + 0] = X[n];
    gin[(size_t)i * 3 + 1] = X[NN + n];
    gin[(size_t)i * 3 + 2] = X[2*NN + n];
}

// -------- m[b,o,j] = sum_n H[row,o] * G[row,j]  (256x16 per batch) --------
__global__ void reduce_m_kernel(const float* __restrict__ H, const float* __restrict__ G,
                                float* __restrict__ m) {
    __shared__ float red[256];
    int blk = blockIdx.x;
    int j = blk & 15, o = (blk >> 4) & 255, b = blk >> 12;
    int tid = threadIdx.x;
    float acc = 0.f;
    for (int n = tid; n < NN; n += 256) {
        size_t row = (size_t)b * NN + n;
        acc += H[row * 256 + o] * G[row * 16 + j];
    }
    red[tid] = acc; __syncthreads();
    for (int s = 128; s > 0; s >>= 1) { if (tid < s) red[tid] += red[tid + s]; __syncthreads(); }
    if (tid == 0) m[(size_t)b * 4096 + o * 16 + j] = red[0];
}

__global__ void final_linear_kernel(const float* __restrict__ m, const float* __restrict__ Wl,
                                    const float* __restrict__ lb, float* __restrict__ y0) {
    __shared__ float red[256];
    int b = blockIdx.x >> 8, o = blockIdx.x & 255, tid = threadIdx.x;
    float acc = 0.f;
    for (int k = tid; k < 4096; k += 256) acc += m[(size_t)b * 4096 + k] * Wl[(size_t)o * 4096 + k];
    red[tid] = acc; __syncthreads();
    for (int s = 128; s > 0; s >>= 1) { if (tid < s) red[tid] += red[tid + s]; __syncthreads(); }
    if (tid == 0) y0[b * 256 + o] = red[0] + lb[o];
}

// final BN over B=2 values per channel, relu, write outputs (6 zeros + 512 y)
__global__ void final_out_kernel(const float* __restrict__ y0, const float* __restrict__ g,
                                 const float* __restrict__ bb, float* __restrict__ d_out) {
    int o = threadIdx.x;
    if (o < 6) d_out[o] = 0.f;                 // new_xyz = zeros(B,3,1)
    float v0 = y0[o], v1 = y0[256 + o];
    float mu = 0.5f * (v0 + v1);
    float var = 0.5f * ((v0 - mu) * (v0 - mu) + (v1 - mu) * (v1 - mu));
    float r = rsqrtf(var + 1e-5f);
    d_out[6 + o]       = fmaxf((v0 - mu) * r * g[o] + bb[o], 0.f);
    d_out[6 + 256 + o] = fmaxf((v1 - mu) * r * g[o] + bb[o], 0.f);
}

// ---------------- host-side orchestration ----------------
extern "C" void kernel_launch(void* const* d_in, const int* in_sizes, int n_in,
                              void* d_out, int out_size, void* d_ws, size_t ws_size,
                              hipStream_t stream) {
    (void)in_sizes; (void)n_in; (void)out_size; (void)ws_size;
    const float* xyz1   = (const float*)d_in[0];
    const float* xyz2   = (const float*)d_in[1];
    const float* points2= (const float*)d_in[3];
    const float* W1  = (const float*)d_in[4];  const float* cb1 = (const float*)d_in[5];
    const float* g1  = (const float*)d_in[6];  const float* b1  = (const float*)d_in[7];
    const float* W2  = (const float*)d_in[8];  const float* cb2 = (const float*)d_in[9];
    const float* g2  = (const float*)d_in[10]; const float* b2  = (const float*)d_in[11];
    const float* dW[3]  = {(const float*)d_in[12], (const float*)d_in[16], (const float*)d_in[20]};
    const float* dcb[3] = {(const float*)d_in[13], (const float*)d_in[17], (const float*)d_in[21]};
    const float* dg[3]  = {(const float*)d_in[14], (const float*)d_in[18], (const float*)d_in[22]};
    const float* db[3]  = {(const float*)d_in[15], (const float*)d_in[19], (const float*)d_in[23]};
    const float* wW[3]  = {(const float*)d_in[24], (const float*)d_in[28], (const float*)d_in[32]};
    const float* wcb[3] = {(const float*)d_in[25], (const float*)d_in[29], (const float*)d_in[33]};
    const float* wg[3]  = {(const float*)d_in[26], (const float*)d_in[30], (const float*)d_in[34]};
    const float* wb[3]  = {(const float*)d_in[27], (const float*)d_in[31], (const float*)d_in[35]};
    const float* Wl  = (const float*)d_in[36]; const float* lb   = (const float*)d_in[37];
    const float* blg = (const float*)d_in[38]; const float* blb  = (const float*)d_in[39];

    char* base = (char*)d_ws;
    size_t off = 0;
    auto alloc = [&](size_t bytes) -> char* {
        char* p = base + off;
        off = (off + bytes + 255) & ~(size_t)255;
        return p;
    };
    unsigned short* W1bf = (unsigned short*)alloc((size_t)O1 * K1P * 2);
    unsigned short* W2bf = (unsigned short*)alloc((size_t)O2 * K2 * 2);
    unsigned short* Xin  = (unsigned short*)alloc((size_t)ROWS * K1P * 2);
    float* H   = (float*)alloc((size_t)ROWS * 256 * 4);   // H1, then reused for H2/Hf
    unsigned short* Hb = (unsigned short*)alloc((size_t)ROWS * 256 * 2);
    float* invd = (float*)alloc((size_t)ROWS * 4);
    float* maxv = (float*)alloc(2 * 4);
    float* mean = (float*)alloc(256 * 4);
    float* rstd = (float*)alloc(256 * 4);
    float* s0 = (float*)alloc((size_t)ROWS * 4);
    float* s1 = (float*)alloc((size_t)ROWS * 16 * 4);
    float* s2 = (float*)alloc((size_t)ROWS * 8 * 4);
    float* s3 = (float*)alloc((size_t)ROWS * 4);
    float* gin = (float*)alloc((size_t)ROWS * 3 * 4);
    float* ga  = (float*)alloc((size_t)ROWS * 8 * 4);
    float* gb2 = (float*)alloc((size_t)ROWS * 8 * 4);
    float* gc  = (float*)alloc((size_t)ROWS * 16 * 4);
    float* mbuf = (float*)alloc((size_t)BB * 4096 * 4);
    float* y0   = (float*)alloc((size_t)BB * 256 * 4);

    // 1) weights -> bf16 (padded)
    wconv_kernel<<<(O1 * K1P + 255) / 256, 256, 0, stream>>>(W1, W1bf, O1, C1, K1P);
    wconv_kernel<<<(O2 * K2 + 255) / 256, 256, 0, stream>>>(W2, W2bf, O2, K2, K2);

    // 2) knn interp -> Xin (bf16, ROWS x 288)
    knn_build_kernel<<<BB * NN / 256, 256, 0, stream>>>(xyz1, xyz2, points2, Xin);

    // 3) density branch front-end
    density_kernel<<<BB * NN / 256, 256, 0, stream>>>(xyz1, invd);
    batch_max_kernel<<<BB, 256, 0, stream>>>(invd, maxv);
    scale_kernel<<<ROWS / 256, 256, 0, stream>>>(invd, maxv, s0);

    // 4) MLP layer 1 (TDM + WMMA): H = Xin * W1^T + cb1
    gemm_bf16_wmma<<<(ROWS / 512) * (O1 / 16), 256, 0, stream>>>(Xin, W1bf, cb1, H, ROWS, K1P, O1);
    stats_kernel<<<256, 256, 0, stream>>>(H, ROWS, 256, mean, rstd);
    bn_relu_kernel<<<ROWS * 256 / 256, 256, 0, stream>>>(H, mean, rstd, g1, b1, ROWS, 256, 1, Hb, nullptr);

    // 5) MLP layer 2 (TDM + WMMA): H = Hb * W2^T + cb2   (reuses H buffer)
    gemm_bf16_wmma<<<(ROWS / 512) * (O2 / 16), 256, 0, stream>>>(Hb, W2bf, cb2, H, ROWS, K2, O2);

    // 6) density MLP branch: (16,1)->(8,16)->(1,8), each conv+bn+relu
    conv_small_kernel<<<ROWS * 16 / 256, 256, 0, stream>>>(s0, dW[0], dcb[0], s1, ROWS, 1, 16);
    stats_kernel<<<16, 256, 0, stream>>>(s1, ROWS, 16, mean, rstd);
    bn_relu_kernel<<<ROWS * 16 / 256, 256, 0, stream>>>(s1, mean, rstd, dg[0], db[0], ROWS, 16, 0, nullptr, nullptr);
    conv_small_kernel<<<ROWS * 8 / 256, 256, 0, stream>>>(s1, dW[1], dcb[1], s2, ROWS, 16, 8);
    stats_kernel<<<8, 256, 0, stream>>>(s2, ROWS, 8, mean, rstd);
    bn_relu_kernel<<<ROWS * 8 / 256, 256, 0, stream>>>(s2, mean, rstd, dg[1], db[1], ROWS, 8, 0, nullptr, nullptr);
    conv_small_kernel<<<ROWS / 256, 256, 0, stream>>>(s2, dW[2], dcb[2], s3, ROWS, 8, 1);
    stats_kernel<<<1, 256, 0, stream>>>(s3, ROWS, 1, mean, rstd);
    bn_relu_kernel<<<ROWS / 256, 256, 0, stream>>>(s3, mean, rstd, dg[2], db[2], ROWS, 1, 0, nullptr, nullptr);

    // 7) BN+ReLU on H2 fused with density-scale multiply -> Hf (in place)
    stats_kernel<<<256, 256, 0, stream>>>(H, ROWS, 256, mean, rstd);
    bn_relu_kernel<<<ROWS * 256 / 256, 256, 0, stream>>>(H, mean, rstd, g2, b2, ROWS, 256, 2, nullptr, s3);

    // 8) weight branch: xyz -> (8,3)->(8,8)->(16,8)
    pack_xyz_kernel<<<ROWS / 256, 256, 0, stream>>>(xyz1, gin);
    conv_small_kernel<<<ROWS * 8 / 256, 256, 0, stream>>>(gin, wW[0], wcb[0], ga, ROWS, 3, 8);
    stats_kernel<<<8, 256, 0, stream>>>(ga, ROWS, 8, mean, rstd);
    bn_relu_kernel<<<ROWS * 8 / 256, 256, 0, stream>>>(ga, mean, rstd, wg[0], wb[0], ROWS, 8, 0, nullptr, nullptr);
    conv_small_kernel<<<ROWS * 8 / 256, 256, 0, stream>>>(ga, wW[1], wcb[1], gb2, ROWS, 8, 8);
    stats_kernel<<<8, 256, 0, stream>>>(gb2, ROWS, 8, mean, rstd);
    bn_relu_kernel<<<ROWS * 8 / 256, 256, 0, stream>>>(gb2, mean, rstd, wg[1], wb[1], ROWS, 8, 0, nullptr, nullptr);
    conv_small_kernel<<<ROWS * 16 / 256, 256, 0, stream>>>(gb2, wW[2], wcb[2], gc, ROWS, 8, 16);
    stats_kernel<<<16, 256, 0, stream>>>(gc, ROWS, 16, mean, rstd);
    bn_relu_kernel<<<ROWS * 16 / 256, 256, 0, stream>>>(gc, mean, rstd, wg[2], wb[2], ROWS, 16, 0, nullptr, nullptr);

    // 9) m = H^T G per batch (256x16), then linear + final BN
    reduce_m_kernel<<<BB * 256 * 16, 256, 0, stream>>>(H, gc, mbuf);
    final_linear_kernel<<<BB * 256, 256, 0, stream>>>(mbuf, Wl, lb, y0);
    final_out_kernel<<<1, 256, 0, stream>>>(y0, blg, blb, (float*)d_out);
}
